// Projection_LSTM_74861279969649
// MI455X (gfx1250) — matrix operand order repaired and checked by reading, not compiled, not run on hardware
//
#include <hip/hip_runtime.h>
#include <stdint.h>

#define SEQ   256
#define BATCH 32
#define EMB   25
#define HID   4096

typedef __attribute__((ext_vector_type(16))) __bf16 v16bf;
typedef __attribute__((ext_vector_type(8)))  float  v8f;

union Frag16 { v16bf v; uint4 q[2]; };

__device__ __forceinline__ uint16_t f32_to_bf16(float f) {
    union { float f; uint32_t u; } c; c.f = f;
    uint32_t u = c.u;
    u += 0x7FFFu + ((u >> 16) & 1u);   // round-to-nearest-even
    return (uint16_t)(u >> 16);
}

// Zero the h ping-pong buffers (h0 = 0) and the barrier counter.
__global__ void rnn_init_kernel(uint16_t* hbufs, unsigned* ctr, int n) {
    int i = blockIdx.x * blockDim.x + threadIdx.x;
    if (i == 0) *ctr = 0u;
    int stride = gridDim.x * blockDim.x;
    for (; i < n; i += stride) hbufs[i] = 0;
}

// W_hh fp32 -> bf16 (L2-resident 32MB working copy).
__global__ void rnn_convert_w_kernel(const float* __restrict__ w,
                                     uint16_t* __restrict__ o, int n) {
    int i = blockIdx.x * blockDim.x + threadIdx.x;
    int stride = gridDim.x * blockDim.x;
    for (; i < n; i += stride) o[i] = f32_to_bf16(w[i]);
}

// Persistent scan kernel: grid = 128 blocks x 128 threads (4 waves).
// Wave (w) owns the 16x16 output tile at rows mb=(w&1)*16 of h (batch dim)
// and cols nb = blockIdx.x*32 + (w>>1)*16 of the hidden dim.
// Per step: C += A(h tile, bf16) x B(W_hh tile, bf16) over K=4096 via
// v_wmma_f32_16x16x32_bf16, then epilogue adds the K=25 input projection,
// biases, tanh, writes h_next (bf16) and the batch-31 row to d_out (fp32).
// Steps are separated by a device-wide monotonic atomic barrier.
__global__ void __launch_bounds__(128)
rnn_scan_kernel(const float* __restrict__ x,      // [S,B,E] fp32
                const float* __restrict__ wih,    // [H,E]   fp32
                const float* __restrict__ bih,    // [H]
                const float* __restrict__ bhh,    // [H]
                const uint16_t* __restrict__ Wb,  // [H,H]   bf16 (W_hh)
                uint16_t* __restrict__ h0,        // [B,H]   bf16 ping
                uint16_t* __restrict__ h1,        // [B,H]   bf16 pong
                float* __restrict__ out,          // [S,H]   fp32
                unsigned* __restrict__ ctr)
{
    const int lane = threadIdx.x & 31;
    const int wave = threadIdx.x >> 5;
    const int mb   = (wave & 1) << 4;                      // batch-tile base
    const int nb   = (blockIdx.x << 5) + ((wave >> 1) << 4); // hidden-tile base
    const int half = lane >> 4;                            // lane 0-15 vs 16-31
    const int l15  = lane & 15;
    const int n    = nb + l15;                             // this lane's output col

    // A-fragment (16-bit A 16x32 layout): row = mb + l15,
    // regs v0..v3 hold K = k + akoff + 0..7, v4..v7 hold K = k + akoff + 16..23.
    const int am    = mb + l15;
    const int akoff = half ? 8 : 0;
    // B-fragment (16-bit B 32x16 layout): col = l15,
    // regs hold K = k + bkoff + 0..7 and k + bkoff + 16..23 (same order as A).
    const int bkoff = half ? 8 : 0;

    const uint16_t* wrow = Wb + (size_t)n * HID;
    const float bias = bih[n] + bhh[n];

    // Cache this lane's W_ih row (25 floats) in registers for the epilogue.
    float wihreg[EMB];
#pragma unroll
    for (int e = 0; e < EMB; ++e) wihreg[e] = wih[n * EMB + e];

    const unsigned nblk = gridDim.x;

    for (int s = 0; s < SEQ; ++s) {
        const uint16_t* hprev = (s & 1) ? h1 : h0;
        uint16_t*       hnext = (s & 1) ? h0 : h1;
        const uint16_t* arow  = hprev + (size_t)am * HID;

        v8f c = {};
#pragma unroll 2
        for (int k = 0; k < HID; k += 64) {
            Frag16 a0, a1, b0, b1;
            a0.q[0] = *(const uint4*)(arow + k + akoff);
            a0.q[1] = *(const uint4*)(arow + k + akoff + 16);
            a1.q[0] = *(const uint4*)(arow + k + 32 + akoff);
            a1.q[1] = *(const uint4*)(arow + k + 32 + akoff + 16);
            b0.q[0] = *(const uint4*)(wrow + k + bkoff);
            b0.q[1] = *(const uint4*)(wrow + k + bkoff + 16);
            b1.q[0] = *(const uint4*)(wrow + k + 32 + bkoff);
            b1.q[1] = *(const uint4*)(wrow + k + 32 + bkoff + 16);
            c = __builtin_amdgcn_wmma_f32_16x16x32_bf16(
                    false, a0.v, false, b0.v, (short)0, c, false, false);
            c = __builtin_amdgcn_wmma_f32_16x16x32_bf16(
                    false, a1.v, false, b1.v, (short)0, c, false, false);
        }

        // Epilogue: C reg r holds (row mb + r) for lanes 0-15, (mb + 8 + r)
        // for lanes 16-31, column n. Add input projection + biases, tanh.
        const int sB = s * BATCH;
#pragma unroll
        for (int r = 0; r < 8; ++r) {
            const int b = mb + (half ? (8 + r) : r);
            const float* xr = x + (size_t)(sB + b) * EMB;
            float acc = c[r] + bias;
#pragma unroll
            for (int e = 0; e < EMB; ++e) acc = fmaf(xr[e], wihreg[e], acc);
            const float hv = tanhf(acc);
            hnext[(size_t)b * HID + n] = f32_to_bf16(hv);
            if (b == BATCH - 1)                  // outputs[:, -1, :]
                out[(size_t)s * HID + n] = hv;
        }

        // Device-wide barrier (monotonic counter; reset by init kernel).
        __threadfence();
        __syncthreads();
        if (threadIdx.x == 0) {
            __hip_atomic_fetch_add(ctr, 1u, __ATOMIC_RELEASE,
                                   __HIP_MEMORY_SCOPE_AGENT);
            const unsigned target = (unsigned)(s + 1) * nblk;
            while (__hip_atomic_load(ctr, __ATOMIC_ACQUIRE,
                                     __HIP_MEMORY_SCOPE_AGENT) < target)
                __builtin_amdgcn_s_sleep(2);
        }
        __syncthreads();
        __threadfence();   // acquire: invalidate WGP$ before reading new h
    }
}

extern "C" void kernel_launch(void* const* d_in, const int* in_sizes, int n_in,
                              void* d_out, int out_size, void* d_ws, size_t ws_size,
                              hipStream_t stream) {
    const float* x   = (const float*)d_in[0];   // [S,B,E]
    const float* Wih = (const float*)d_in[1];   // [H,E]
    const float* Whh = (const float*)d_in[2];   // [H,H]
    const float* bih = (const float*)d_in[3];   // [H]
    const float* bhh = (const float*)d_in[4];   // [H]
    float* out = (float*)d_out;                 // [S,H]

    uint8_t* ws = (uint8_t*)d_ws;
    uint16_t* Wb  = (uint16_t*)ws;                               // H*H bf16 = 32MB
    uint16_t* h0  = (uint16_t*)(ws + (size_t)HID * HID * 2);     // B*H bf16
    uint16_t* h1  = h0 + (size_t)BATCH * HID;                    // B*H bf16
    unsigned* ctr = (unsigned*)(ws + (size_t)HID * HID * 2
                                   + 2 * (size_t)BATCH * HID * 2);

    rnn_init_kernel<<<64, 256, 0, stream>>>(h0, ctr, 2 * BATCH * HID);
    rnn_convert_w_kernel<<<2048, 256, 0, stream>>>(Whh, Wb, HID * HID);
    rnn_scan_kernel<<<128, 128, 0, stream>>>(x, Wih, bih, bhh, Wb, h0, h1,
                                             out, ctr);
}
